// MPNNMultidimFullAttention_12515534701093
// MI455X (gfx1250) — matrix operand order repaired and checked by reading, compiled, not run on hardware
//
#include <hip/hip_runtime.h>
#include <hip/hip_bf16.h>

#define LNUM 3
#define HNUM 8
#define DDIM 128
#define DKDIM 16
#define FFDIM 512
#define BNUM 32
#define NSEQ 512
#define FIN 16
#define OUTD 3
#define MROWS (BNUM * NSEQ)   // 16384
#define ATT_SCALE 0.25f
#define NEGBIG (-1e30f)

typedef __attribute__((ext_vector_type(16))) _Float16 v16h;
typedef __attribute__((ext_vector_type(8)))  float    v8f;

union AF { v16h v; unsigned u[8]; _Float16 h[16]; };

// ---- WMMA fragment loaders (contiguous k order k = 16*(lane>>4) + s, wave32) ----

// A: 16x32 f16, row-major source, row = lane&15,
// lanes 0-15: K = 0..15 ; lanes 16-31: K = 16..31
__device__ __forceinline__ v16h load_a_frag(const _Float16* A, int ldaH, int row, int k0, int lane) {
    const unsigned* a32 = (const unsigned*)(A + (size_t)row * ldaH + k0);
    int g4 = (lane >> 4) << 3;
    AF f;
#pragma unroll
    for (int j = 0; j < 8; ++j) f.u[j] = a32[g4 + j];
    return f.v;
}

// B: 32x16 f16 from B^T storage (BT[n*K + k]); col = lane&15, K = 16*(lane>>4)+j
__device__ __forceinline__ v16h load_bt_frag(const _Float16* BT, int K, int col, int k0, int lane) {
    const unsigned* b32 = (const unsigned*)(BT + (size_t)col * K + k0 + ((lane >> 4) << 4));
    AF f;
#pragma unroll
    for (int j = 0; j < 8; ++j) f.u[j] = b32[j];
    return f.v;
}

// 16x64 output tile per wave: one shared A fragment feeds 4 WMMAs
__device__ __forceinline__ void gemm_tile4(const _Float16* A, int ldaH, int row0,
                                           const _Float16* BT, int K, int col0, int lane,
                                           v8f acc[4]) {
    int row = row0 + (lane & 15);
    int col = col0 + (lane & 15);
    for (int k0 = 0; k0 < K; k0 += 32) {
        v16h a = load_a_frag(A, ldaH, row, k0, lane);
#pragma unroll
        for (int t = 0; t < 4; ++t) {
            v16h b = load_bt_frag(BT, K, col + 16 * t, k0, lane);
            acc[t] = __builtin_amdgcn_wmma_f32_16x16x32_f16(false, a, false, b, (short)0, acc[t], false, false);
        }
    }
}

// ---- weight preparation ----

// Wq/Wk/Wv: [L,H,D,DK] -> BT[l][c*D + d] with c = h*16+dk (f16)
__global__ void k_prep_qkv(const float* __restrict__ W, _Float16* __restrict__ BT) {
    int tid = blockIdx.x * blockDim.x + threadIdx.x;
    if (tid >= LNUM * DDIM * DDIM) return;
    int l = tid / (DDIM * DDIM);
    int rem = tid - l * (DDIM * DDIM);
    int c = rem >> 7;
    int d = rem & 127;
    float v = W[((size_t)l * HNUM + (c >> 4)) * DDIM * DKDIM + d * DKDIM + (c & 15)];
    BT[(size_t)l * DDIM * DDIM + (size_t)c * DDIM + d] = (_Float16)v;
}

// generic per-layer transpose + f16 convert: in [L,R,C] -> out[l][c*R + r]
__global__ void k_transpose(const float* __restrict__ W, _Float16* __restrict__ BT, int R, int C) {
    int tid = blockIdx.x * blockDim.x + threadIdx.x;
    int total = LNUM * R * C;
    if (tid >= total) return;
    int l = tid / (R * C);
    int rem = tid - l * (R * C);
    int r = rem / C;
    int c = rem - r * C;
    BT[(size_t)l * R * C + (size_t)c * R + r] = (_Float16)W[(size_t)l * R * C + (size_t)r * C + c];
}

// ---- embedding: h = relu(x @ We) ----
__global__ void k_embed(const float* __restrict__ x, const float* __restrict__ We,
                        float* __restrict__ h32, _Float16* __restrict__ h16) {
    int tid = blockIdx.x * blockDim.x + threadIdx.x;   // MROWS*DDIM threads
    int i = tid >> 7;
    int j = tid & 127;
    float s = 0.f;
#pragma unroll
    for (int k = 0; k < FIN; ++k) s += x[i * FIN + k] * We[k * DDIM + j];
    s = fmaxf(s, 0.f);
    h32[tid] = s;
    h16[tid] = (_Float16)s;
}

// ---- QKV projection: [16384,128]x[128,128], one 16x64 tile per wave ----
__global__ void __launch_bounds__(128) k_qkv(const _Float16* __restrict__ h16,
                      const _Float16* __restrict__ WqT, const _Float16* __restrict__ WkT,
                      const _Float16* __restrict__ WvT,
                      _Float16* __restrict__ Qd, _Float16* __restrict__ Kd,
                      _Float16* __restrict__ Vt, int l) {
    int lane = threadIdx.x & 31;
    int wave = threadIdx.x >> 5;
    int tile = blockIdx.x * 4 + wave;        // 2048 wave-tiles (1024 row x 2 supercol)
    int mt = tile >> 1, nt = tile & 1;
    int row0 = mt << 4, col0 = nt << 6;
    const _Float16* BT;
    _Float16* outq = nullptr;
    int isV = 0;
    if (blockIdx.y == 0)      { BT = WqT; outq = Qd; }
    else if (blockIdx.y == 1) { BT = WkT; outq = Kd; }
    else                      { BT = WvT; isV = 1; }
    BT += (size_t)l * DDIM * DDIM;
    v8f acc[4] = {};
    gemm_tile4(h16, DDIM, row0, BT, DDIM, col0, lane, acc);
    int g = lane >> 4, n = lane & 15;
#pragma unroll
    for (int t = 0; t < 4; ++t) {
#pragma unroll
        for (int v = 0; v < 8; ++v) {
            int r = row0 + v + 8 * g;
            int c = col0 + 16 * t + n;
            int b = r >> 9, ns = r & (NSEQ - 1);
            int hh = c >> 4, dk = c & 15;
            _Float16 val = (_Float16)acc[t][v];
            if (isV) Vt[(((size_t)b * HNUM + hh) * DKDIM + dk) * NSEQ + ns] = val;
            else     outq[(((size_t)b * HNUM + hh) * NSEQ + ns) * DKDIM + dk] = val;
        }
    }
}

// ---- flash attention: one wave per (b, head, 16-query tile), 32 keys/iter ----
__global__ void __launch_bounds__(128) k_attn(const _Float16* __restrict__ Q,
                                              const _Float16* __restrict__ Kd,
                                              const _Float16* __restrict__ Vt,
                                              const int* __restrict__ mask,
                                              _Float16* __restrict__ O) {
    __shared__ __align__(16) _Float16 plds[4][16][32];
    int lane = threadIdx.x & 31;
    int wave = threadIdx.x >> 5;
    int idx = blockIdx.x * 4 + wave;         // 8192 wave-tiles
    int qt = idx & 31;  idx >>= 5;
    int hh = idx & 7;   idx >>= 3;
    int b  = idx;
    const _Float16* Qbh = Q  + ((size_t)b * HNUM + hh) * NSEQ * DKDIM;
    const _Float16* Kbh = Kd + ((size_t)b * HNUM + hh) * NSEQ * DKDIM;
    const _Float16* Vbh = Vt + ((size_t)b * HNUM + hh) * DKDIM * NSEQ;
    const int* mb = mask + (size_t)b * NSEQ * NSEQ;
    int q0 = qt << 4;
    int g = lane >> 4, n = lane & 15;

    // Q A-fragment, K-dim padded 16->32 with zeros
    AF qa;
    {
        const unsigned* q32 = (const unsigned*)(Qbh + (size_t)(q0 + n) * DKDIM);
#pragma unroll
        for (int j = 0; j < 8; ++j) qa.u[j] = (g == 0) ? q32[j] : 0u;
    }

    float mstate[8], lstate[8];
    v8f oacc = {};
#pragma unroll
    for (int v = 0; v < 8; ++v) { mstate[v] = NEGBIG; lstate[v] = 0.f; }

    for (int kt = 0; kt < NSEQ / 32; ++kt) {
        int keybase = kt << 5;
        // two K^T B-fragments (only K-dim 0..15 valid -> upper half zero)
        AF kb0, kb1;
        if (g == 0) {
            const unsigned* ka = (const unsigned*)(Kbh + (size_t)(keybase + n) * DKDIM);
            const unsigned* kbp = (const unsigned*)(Kbh + (size_t)(keybase + 16 + n) * DKDIM);
#pragma unroll
            for (int j = 0; j < 8; ++j) { kb0.u[j] = ka[j]; kb1.u[j] = kbp[j]; }
        } else {
#pragma unroll
            for (int j = 0; j < 8; ++j) { kb0.u[j] = 0u; kb1.u[j] = 0u; }
        }
        v8f sa8 = {}, sb8 = {};
        sa8 = __builtin_amdgcn_wmma_f32_16x16x32_f16(false, qa.v, false, kb0.v, (short)0, sa8, false, false);
        sb8 = __builtin_amdgcn_wmma_f32_16x16x32_f16(false, qa.v, false, kb1.v, (short)0, sb8, false, false);

        float pva[8], pvb[8];
#pragma unroll
        for (int v = 0; v < 8; ++v) {
            int qrow = q0 + v + 8 * g;
            const int* mrow = mb + (size_t)qrow * NSEQ + keybase;
            float sa = sa8[v] * ATT_SCALE;
            float sb = sb8[v] * ATT_SCALE;
            if (mrow[n] == 0)      sa = NEGBIG;
            if (mrow[16 + n] == 0) sb = NEGBIG;
            float rm = fmaxf(sa, sb);
            rm = fmaxf(rm, __shfl_xor(rm, 1, 32));
            rm = fmaxf(rm, __shfl_xor(rm, 2, 32));
            rm = fmaxf(rm, __shfl_xor(rm, 4, 32));
            rm = fmaxf(rm, __shfl_xor(rm, 8, 32));
            float mnew = fmaxf(mstate[v], rm);
            float pa   = __expf(sa - mnew);
            float pb   = __expf(sb - mnew);
            float corr = __expf(mstate[v] - mnew);
            float ps = pa + pb;
            ps += __shfl_xor(ps, 1, 32);
            ps += __shfl_xor(ps, 2, 32);
            ps += __shfl_xor(ps, 4, 32);
            ps += __shfl_xor(ps, 8, 32);
            lstate[v] = lstate[v] * corr + ps;
            mstate[v] = mnew;
            oacc[v]   = oacc[v] * corr;
            pva[v] = pa;
            pvb[v] = pb;
        }
        __syncthreads();
#pragma unroll
        for (int v = 0; v < 8; ++v) {
            plds[wave][v + 8 * g][n]      = (_Float16)pva[v];
            plds[wave][v + 8 * g][16 + n] = (_Float16)pvb[v];
        }
        __syncthreads();
        // P as full 16x32 A-fragment
        AF pa;
        {
            const unsigned* p32 = (const unsigned*)&plds[wave][n][0];
#pragma unroll
            for (int j = 0; j < 8; ++j) pa.u[j] = p32[8 * g + j];
        }
        // V as full 32x16 B-fragment from transposed V (contiguous along keys)
        AF vb;
        {
            const unsigned* v32 = (const unsigned*)(Vbh + (size_t)n * NSEQ + keybase + (g << 4));
#pragma unroll
            for (int j = 0; j < 8; ++j) vb.u[j] = v32[j];
        }
        oacc = __builtin_amdgcn_wmma_f32_16x16x32_f16(false, pa.v, false, vb.v, (short)0, oacc, false, false);
        __syncthreads();
    }
#pragma unroll
    for (int v = 0; v < 8; ++v) {
        int qrow = q0 + v + 8 * g;
        float val = oacc[v] / lstate[v];
        O[((size_t)b * NSEQ + qrow) * DDIM + hh * DKDIM + n] = (_Float16)val;
    }
}

// ---- h += O @ Wout ----
__global__ void __launch_bounds__(128) k_wout(const _Float16* __restrict__ O16,
                       const _Float16* __restrict__ WoutT,
                       float* __restrict__ h32, _Float16* __restrict__ h16, int l) {
    int lane = threadIdx.x & 31, wave = threadIdx.x >> 5;
    int tile = blockIdx.x * 4 + wave;        // 2048 wave-tiles
    int mt = tile >> 1, nt = tile & 1;
    int row0 = mt << 4, col0 = nt << 6;
    v8f acc[4] = {};
    gemm_tile4(O16, DDIM, row0, WoutT + (size_t)l * DDIM * DDIM, DDIM, col0, lane, acc);
    int g = lane >> 4, n = lane & 15;
#pragma unroll
    for (int t = 0; t < 4; ++t) {
#pragma unroll
        for (int v = 0; v < 8; ++v) {
            size_t idx2 = (size_t)(row0 + v + 8 * g) * DDIM + col0 + 16 * t + n;
            float val = h32[idx2] + acc[t][v];
            h32[idx2] = val;
            h16[idx2] = (_Float16)val;
        }
    }
}

// ---- t = relu(h @ Wff1 + bff1) ----
__global__ void __launch_bounds__(128) k_ff1(const _Float16* __restrict__ h16,
                      const _Float16* __restrict__ W1T, const float* __restrict__ bff1,
                      _Float16* __restrict__ t16, int l) {
    int lane = threadIdx.x & 31, wave = threadIdx.x >> 5;
    int tile = blockIdx.x * 4 + wave;        // 8192 wave-tiles (1024 x 8)
    int mt = tile >> 3, nt = tile & 7;
    int row0 = mt << 4, col0 = nt << 6;
    v8f acc[4] = {};
    gemm_tile4(h16, DDIM, row0, W1T + (size_t)l * FFDIM * DDIM, DDIM, col0, lane, acc);
    int g = lane >> 4, n = lane & 15;
#pragma unroll
    for (int t = 0; t < 4; ++t) {
        float bias = bff1[l * FFDIM + col0 + 16 * t + n];
#pragma unroll
        for (int v = 0; v < 8; ++v) {
            float val = fmaxf(acc[t][v] + bias, 0.f);
            t16[(size_t)(row0 + v + 8 * g) * FFDIM + col0 + 16 * t + n] = (_Float16)val;
        }
    }
}

// ---- h += t @ Wff2 + bff2 ----
__global__ void __launch_bounds__(128) k_ff2(const _Float16* __restrict__ t16,
                      const _Float16* __restrict__ W2T, const float* __restrict__ bff2,
                      float* __restrict__ h32, _Float16* __restrict__ h16, int l) {
    int lane = threadIdx.x & 31, wave = threadIdx.x >> 5;
    int tile = blockIdx.x * 4 + wave;        // 2048 wave-tiles
    int mt = tile >> 1, nt = tile & 1;
    int row0 = mt << 4, col0 = nt << 6;
    v8f acc[4] = {};
    gemm_tile4(t16, FFDIM, row0, W2T + (size_t)l * DDIM * FFDIM, FFDIM, col0, lane, acc);
    int g = lane >> 4, n = lane & 15;
#pragma unroll
    for (int t = 0; t < 4; ++t) {
        float bias = bff2[l * DDIM + col0 + 16 * t + n];
#pragma unroll
        for (int v = 0; v < 8; ++v) {
            size_t idx2 = (size_t)(row0 + v + 8 * g) * DDIM + col0 + 16 * t + n;
            float val = h32[idx2] + acc[t][v] + bias;
            h32[idx2] = val;
            h16[idx2] = (_Float16)val;
        }
    }
}

// ---- readout ----
__global__ void k_pool(const float* __restrict__ h32, float* __restrict__ pooled) {
    int tid = blockIdx.x * blockDim.x + threadIdx.x;   // 4096
    int b = tid >> 7, d = tid & 127;
    float s = 0.f;
    for (int nn = 0; nn < NSEQ; ++nn) s += h32[((size_t)b * NSEQ + nn) * DDIM + d];
    pooled[tid] = s * (1.0f / NSEQ);
}

__global__ void k_hp(const float* __restrict__ pooled, const float* __restrict__ Wp,
                     float* __restrict__ hp) {
    int tid = blockIdx.x * blockDim.x + threadIdx.x;   // 4096
    int b = tid >> 7, d2 = tid & 127;
    float s = 0.f;
    for (int d = 0; d < DDIM; ++d) s += pooled[b * DDIM + d] * Wp[d * DDIM + d2];
    hp[tid] = s;
}

__global__ void k_out(const float* __restrict__ hp, const float* __restrict__ h32,
                      const float* __restrict__ Wr, const float* __restrict__ br,
                      float* __restrict__ out) {
    int tid = blockIdx.x * blockDim.x + threadIdx.x;   // B*N*OUTD = 49152
    int b = tid / (NSEQ * OUTD);
    int rem = tid - b * NSEQ * OUTD;
    int nn = rem / OUTD;
    int o = rem - nn * OUTD;
    float s = br[o];
    for (int j = 0; j < DDIM; ++j)
        s += fmaxf(hp[b * DDIM + j], 0.f) * Wr[j * OUTD + o];
    for (int j = 0; j < DDIM; ++j)
        s += fmaxf(h32[((size_t)b * NSEQ + nn) * DDIM + j], 0.f) * Wr[(DDIM + j) * OUTD + o];
    out[tid] = s;
}

extern "C" void kernel_launch(void* const* d_in, const int* in_sizes, int n_in,
                              void* d_out, int out_size, void* d_ws, size_t ws_size,
                              hipStream_t stream) {
    const float* x    = (const float*)d_in[0];
    const int*   mask = (const int*)d_in[1];
    const float* We   = (const float*)d_in[2];
    const float* Wq   = (const float*)d_in[3];
    const float* Wk   = (const float*)d_in[4];
    const float* Wv   = (const float*)d_in[5];
    const float* Wout = (const float*)d_in[6];
    const float* Wff1 = (const float*)d_in[7];
    const float* bff1 = (const float*)d_in[8];
    const float* Wff2 = (const float*)d_in[9];
    const float* bff2 = (const float*)d_in[10];
    const float* Wp   = (const float*)d_in[11];
    const float* Wr   = (const float*)d_in[12];
    const float* br   = (const float*)d_in[13];
    float* out = (float*)d_out;

    char* w = (char*)d_ws;
    size_t off = 0;
    auto alloc = [&](size_t bytes) -> void* {
        void* p = w + off;
        off += (bytes + 255) & ~(size_t)255;
        return p;
    };

    float*    h32  = (float*)   alloc((size_t)MROWS * DDIM * 4);
    _Float16* h16  = (_Float16*)alloc((size_t)MROWS * DDIM * 2);
    _Float16* Q16  = (_Float16*)alloc((size_t)MROWS * DDIM * 2);
    _Float16* K16  = (_Float16*)alloc((size_t)MROWS * DDIM * 2);
    _Float16* V16t = (_Float16*)alloc((size_t)MROWS * DDIM * 2);
    _Float16* O16  = (_Float16*)alloc((size_t)MROWS * DDIM * 2);
    _Float16* t16  = (_Float16*)alloc((size_t)MROWS * FFDIM * 2);
    _Float16* WqT  = (_Float16*)alloc((size_t)LNUM * DDIM * DDIM * 2);
    _Float16* WkT  = (_Float16*)alloc((size_t)LNUM * DDIM * DDIM * 2);
    _Float16* WvT  = (_Float16*)alloc((size_t)LNUM * DDIM * DDIM * 2);
    _Float16* WoT  = (_Float16*)alloc((size_t)LNUM * DDIM * DDIM * 2);
    _Float16* W1T  = (_Float16*)alloc((size_t)LNUM * DDIM * FFDIM * 2);
    _Float16* W2T  = (_Float16*)alloc((size_t)LNUM * FFDIM * DDIM * 2);
    float*    pooled = (float*)alloc(BNUM * DDIM * 4);
    float*    hp     = (float*)alloc(BNUM * DDIM * 4);

    // weight prep
    k_prep_qkv<<<192, 256, 0, stream>>>(Wq, WqT);
    k_prep_qkv<<<192, 256, 0, stream>>>(Wk, WkT);
    k_prep_qkv<<<192, 256, 0, stream>>>(Wv, WvT);
    k_transpose<<<192, 256, 0, stream>>>(Wout, WoT, DDIM, DDIM);    // [DK*H, D] flat == [128,128]
    k_transpose<<<768, 256, 0, stream>>>(Wff1, W1T, DDIM, FFDIM);
    k_transpose<<<768, 256, 0, stream>>>(Wff2, W2T, FFDIM, DDIM);

    // embedding
    k_embed<<<(MROWS * DDIM) / 256, 256, 0, stream>>>(x, We, h32, h16);

    for (int l = 0; l < LNUM; ++l) {
        k_qkv<<<dim3(512, 3, 1), 128, 0, stream>>>(h16, WqT, WkT, WvT, Q16, K16, V16t, l);
        k_attn<<<(BNUM * HNUM * (NSEQ / 16)) / 4, 128, 0, stream>>>(Q16, K16, V16t, mask, O16);
        k_wout<<<512, 128, 0, stream>>>(O16, WoT, h32, h16, l);
        k_ff1<<<2048, 128, 0, stream>>>(h16, W1T, bff1, t16, l);
        k_ff2<<<512, 128, 0, stream>>>(t16, W2T, bff2, h32, h16, l);
    }

    k_pool<<<16, 256, 0, stream>>>(h32, pooled);
    k_hp<<<16, 256, 0, stream>>>(pooled, Wp, hp);
    k_out<<<192, 256, 0, stream>>>(hp, h32, Wr, br, out);
}
